// TemporalGCN_52948356825190
// MI455X (gfx1250) — matrix operand order repaired and checked by reading, not compiled, not run on hardware
//
#include <hip/hip_runtime.h>
#include <cstdint>

typedef unsigned short u16;
typedef __attribute__((ext_vector_type(16))) __bf16 v16bf;
typedef __attribute__((ext_vector_type(8)))  float  v8f;
typedef int v4i __attribute__((vector_size(16)));   // matches builtin param type

#define N_NODES 65536
#define IN_DIM  128
#define HID     256
#define OUT_DIM 128
#define N_EDGES 1048576
#define BATCH   64
#define SEQ_T   1024
#define GATE4   512

#define GLOBAL_AS __attribute__((address_space(1)))
#define LOCAL_AS  __attribute__((address_space(3)))

#if defined(__HIP_DEVICE_COMPILE__) && __has_builtin(__builtin_amdgcn_global_load_async_to_lds_b128)
#define HAVE_ASYNC_LDS 1
#else
#define HAVE_ASYNC_LDS 0
#endif

#if HAVE_ASYNC_LDS
__device__ __forceinline__ void async_cp16(const u16* g, u16* l) {
  __builtin_amdgcn_global_load_async_to_lds_b128(
      (GLOBAL_AS v4i*)(unsigned long long)(uintptr_t)g,   // AS(1): full 64-bit addr
      (LOCAL_AS v4i*)(unsigned)(uintptr_t)l,              // AS(3): low 32 bits = LDS offset
      0, 0);
}
#if __has_builtin(__builtin_amdgcn_s_wait_asynccnt)
#define WAIT_ASYNC(n) __builtin_amdgcn_s_wait_asynccnt(n)
#else
#define WAIT_ASYNC(n) asm volatile("s_wait_asynccnt " #n ::: "memory")
#endif
#endif

// ---------- helpers ----------
__device__ __forceinline__ u16 f2bf(float f) {
  unsigned u = __float_as_uint(f);
  unsigned r = u + 0x7FFFu + ((u >> 16) & 1u);   // round-to-nearest-even
  return (u16)(r >> 16);
}

__device__ __forceinline__ void atomAddF(float* p, float v) {
  unsafeAtomicAdd(p, v);   // native global_atomic_add_f32 on gfx1250
}

// ---------- tiny elementwise kernels ----------
__global__ void k_fill1(float* p, int n) {
  int i = blockIdx.x * blockDim.x + threadIdx.x;
  if (i < n) p[i] = 1.0f;
}

__global__ void k_deg(const int* __restrict__ dst, float* __restrict__ deg, int e) {
  int i = blockIdx.x * blockDim.x + threadIdx.x;
  if (i < e) atomAddF(&deg[dst[i]], 1.0f);
}

__global__ void k_rsqrt_inplace(float* p, int n) {
  int i = blockIdx.x * blockDim.x + threadIdx.x;
  if (i < n) p[i] = rsqrtf(p[i]);
}

__global__ void k_norm(const int* __restrict__ src, const int* __restrict__ dst,
                       const float* __restrict__ dinv, float* __restrict__ nrm, int e) {
  int i = blockIdx.x * blockDim.x + threadIdx.x;
  if (i < e) nrm[i] = dinv[src[i]] * dinv[dst[i]];
}

__global__ void k_f2bf(const float* __restrict__ in, u16* __restrict__ out, int n) {
  int i = blockIdx.x * blockDim.x + threadIdx.x;
  if (i < n) out[i] = f2bf(in[i]);
}

__global__ void k_relubf(const float* __restrict__ in, u16* __restrict__ out, int n) {
  int i = blockIdx.x * blockDim.x + threadIdx.x;
  if (i < n) out[i] = f2bf(fmaxf(in[i], 0.0f));
}

// out[n*K+k] = bf16(in[k*N+n])  -- convert [K,N] weight into [N,K] "Bt" layout
__global__ void k_transpose_bf(const float* __restrict__ in, u16* __restrict__ out,
                               int K, int N) {
  int i = blockIdx.x * blockDim.x + threadIdx.x;
  if (i < K * N) {
    int n = i / K, k = i % K;
    out[i] = f2bf(in[(size_t)k * N + n]);
  }
}

__global__ void k_addbias2(const float* a, const float* b, float* out, int n) {
  int i = blockIdx.x * blockDim.x + threadIdx.x;
  if (i < n) out[i] = a[i] + b[i];
}

// agg[n][f] = bias[f] + h[n][f]*dinv[n]^2   (self-loop + bias, non-atomic init)
__global__ void k_initagg(const float* __restrict__ h, const float* __restrict__ dinv,
                          const float* __restrict__ bias, float* __restrict__ agg, int n) {
  int i = blockIdx.x * blockDim.x + threadIdx.x;
  if (i < n) {
    float d = dinv[i >> 8];
    agg[i] = bias[i & 255] + h[i] * d * d;
  }
}

// one thread per (edge, 4-feature chunk): agg[dst] += h[src]*norm
__global__ void k_scatter(const int* __restrict__ src, const int* __restrict__ dst,
                          const float* __restrict__ nrm, const float* __restrict__ h,
                          float* __restrict__ agg, int e) {
  long long idx = (long long)blockIdx.x * blockDim.x + threadIdx.x;
  int ei = (int)(idx >> 6);
  if (ei >= e) return;
  int f4 = (int)(idx & 63) << 2;
  float w = nrm[ei];
  int s = src[ei], d = dst[ei];
  float4 hv = *(const float4*)&h[(size_t)s * HID + f4];
  float* dp = &agg[(size_t)d * HID + f4];
  atomAddF(dp + 0, hv.x * w);
  atomAddF(dp + 1, hv.y * w);
  atomAddF(dp + 2, hv.z * w);
  atomAddF(dp + 3, hv.w * w);
}

// LayerNorm over 256 feats, one wave per row, write bf16
__global__ __launch_bounds__(256)
void k_layernorm(const float* __restrict__ h, const float* __restrict__ gamma,
                 const float* __restrict__ beta, u16* __restrict__ out, int rows) {
  int wave = threadIdx.x >> 5, lane = threadIdx.x & 31;
  int row = blockIdx.x * 8 + wave;
  if (row >= rows) return;
  const float* rp = &h[(size_t)row * HID];
  float4 v0 = *(const float4*)&rp[lane * 8];
  float4 v1 = *(const float4*)&rp[lane * 8 + 4];
  float s  = v0.x + v0.y + v0.z + v0.w + v1.x + v1.y + v1.z + v1.w;
  float sq = v0.x*v0.x + v0.y*v0.y + v0.z*v0.z + v0.w*v0.w
           + v1.x*v1.x + v1.y*v1.y + v1.z*v1.z + v1.w*v1.w;
  #pragma unroll
  for (int off = 16; off > 0; off >>= 1) {
    s  += __shfl_xor(s,  off, 32);
    sq += __shfl_xor(sq, off, 32);
  }
  float mean = s * (1.0f / HID);
  float var  = sq * (1.0f / HID) - mean * mean;
  float rstd = rsqrtf(var + 1e-5f);
  float xv[8] = {v0.x, v0.y, v0.z, v0.w, v1.x, v1.y, v1.z, v1.w};
  #pragma unroll
  for (int i = 0; i < 8; i++) {
    int c = lane * 8 + i;
    out[(size_t)row * HID + c] = f2bf((xv[i] - mean) * rstd * gamma[c] + beta[c]);
  }
}

// ---------- bf16 WMMA GEMM:  C[M,N] = A[M,K] * Bt[N,K]^T (+bias) ----------
// A, Bt row-major bf16; 128x128 block tile, 8 waves, each wave = 16M x 128N strip.
// Double-buffered LDS tiles filled by async global->LDS copies (ASYNCcnt).
__global__ __launch_bounds__(256)
void gemm_bf16_nt(const u16* __restrict__ A, const u16* __restrict__ Bt,
                  const float* __restrict__ bias, float* __restrict__ C,
                  int M, int N, int K) {
  __shared__ u16 As[2][128][40];   // row stride 80B: 16B aligned, conflict-skewed
  __shared__ u16 Bs[2][128][40];
  const int tid = threadIdx.x;
  const int wave = tid >> 5, lane = tid & 31;
  const int hi = (lane >> 4) & 1, l15 = lane & 15;
  const int bm = blockIdx.x * 128, bn = blockIdx.y * 128;

  const v8f zero = {0.f, 0.f, 0.f, 0.f, 0.f, 0.f, 0.f, 0.f};
  v8f acc[8];
  #pragma unroll
  for (int nt = 0; nt < 8; nt++) acc[nt] = zero;

  const int nk = K >> 5;

#if HAVE_ASYNC_LDS
  auto issue = [&](int kk, int buf) {
    #pragma unroll
    for (int it = 0; it < 2; ++it) {
      int li = tid + it * 256;
      int r = li >> 2, c = (li & 3) * 8;
      async_cp16(A  + (size_t)(bm + r) * K + kk + c, &As[buf][r][c]);
      async_cp16(Bt + (size_t)(bn + r) * K + kk + c, &Bs[buf][r][c]);
    }
  };
  issue(0, 0);
#endif

  for (int i = 0; i < nk; ++i) {
    const int kk = i << 5;
#if HAVE_ASYNC_LDS
    if (i + 1 < nk) {
      issue(kk + 32, (i + 1) & 1);   // stream next tile while computing this one
      WAIT_ASYNC(4);                 // tile i's 4 async copies done (in-order)
    } else {
      WAIT_ASYNC(0);
    }
    __syncthreads();
    const int cur = i & 1;
#else
    #pragma unroll
    for (int it = 0; it < 2; ++it) {
      int li = tid + it * 256;
      int r = li >> 2, c = (li & 3) * 8;
      *(uint4*)&As[0][r][c] = *(const uint4*)(A  + (size_t)(bm + r) * K + kk + c);
      *(uint4*)&Bs[0][r][c] = *(const uint4*)(Bt + (size_t)(bn + r) * K + kk + c);
      if (i + 1 < nk) {              // prefetch next tile into cache
        __builtin_prefetch(A  + (size_t)(bm + r) * K + kk + 32 + c, 0, 3);
        __builtin_prefetch(Bt + (size_t)(bn + r) * K + kk + 32 + c, 0, 3);
      }
    }
    __syncthreads();
    const int cur = 0;
#endif

    union { int i[8]; v16bf v; } a;
    #pragma unroll
    for (int q = 0; q < 8; q++) {                  // A frag: 16x32 bf16 layout
      int k0 = 2 * q + hi * 16;
      a.i[q] = *(const int*)&As[cur][wave * 16 + l15][k0];
    }
    #pragma unroll
    for (int nt = 0; nt < 8; nt++) {
      union { int i[8]; v16bf v; } b;
      #pragma unroll
      for (int q = 0; q < 8; q++) {                // B frag: 32x16 from [N,K] rows
        int k0 = 2 * q + hi * 16;
        b.i[q] = *(const int*)&Bs[cur][nt * 16 + l15][k0];
      }
      acc[nt] = __builtin_amdgcn_wmma_f32_16x16x32_bf16(
          false, a.v, false, b.v, (short)0, acc[nt], false, false);
    }
    __syncthreads();
  }

  #pragma unroll
  for (int nt = 0; nt < 8; nt++) {
    int col = bn + nt * 16 + l15;
    float bv = bias ? bias[col] : 0.0f;
    #pragma unroll
    for (int r = 0; r < 8; r++) {
      int row = bm + wave * 16 + r + 8 * hi;       // C layout: M = r + 8*hi
      C[(size_t)row * N + col] = acc[nt][r] + bv;
    }
  }
}

// ---------- LSTM recurrence (input gates precomputed into `gates`) ----------
// grid = 4 blocks x 16 batch rows. Per step: [16,128] bf16 h @ Whh^T[128,512] via WMMA.
// All Whh B-fragments are hoisted into registers before the time loop (128 VGPRs;
// only 2 waves/SIMD so no pressure). Gates for t+1 are prefetched each step.
__global__ __launch_bounds__(256)
void k_lstm(const float* __restrict__ gates,       // [(B*T), 512] f32, order i,f,g,o
            const u16* __restrict__ Whh,           // [512,128] bf16 (N,K layout)
            u16* __restrict__ ys,                  // [(B*T),128] bf16 or nullptr
            float* __restrict__ hT,                // [B,128] f32 or nullptr
            int T) {
  __shared__ u16   h_s[16][136];                   // bf16 h, padded row
  __shared__ float c_s[16][128];
  __shared__ float gbuf[16][512];
  const int tid = threadIdx.x;
  const int wave = tid >> 5, lane = tid & 31;
  const int hi = (lane >> 4) & 1, l15 = lane & 15;
  const int b0 = blockIdx.x * 16;

  for (int i = tid; i < 16 * 136; i += 256) ((u16*)h_s)[i] = 0;
  for (int i = tid; i < 16 * 128; i += 256) ((float*)c_s)[i] = 0.0f;
  __syncthreads();

  // hoist Whh fragments into registers: bfr[kk][nt], 16 x v16bf
  union Bfr { int i[8]; v16bf v; };
  Bfr bfr[4][4];
  #pragma unroll
  for (int kk = 0; kk < 4; kk++) {
    #pragma unroll
    for (int nt = 0; nt < 4; nt++) {
      int brow = wave * 64 + nt * 16 + l15;
      #pragma unroll
      for (int q = 0; q < 8; q++) {
        int k0 = kk * 32 + 2 * q + hi * 16;
        bfr[kk][nt].i[q] = *(const int*)(Whh + (size_t)brow * 128 + k0);
      }
    }
  }

  for (int t = 0; t < T; ++t) {
    // prefetch next step's gate rows (one 128B line per thread = full 32KB)
    if (t + 1 < T) {
      int prow = tid >> 4, pseg = (tid & 15) * 32;
      __builtin_prefetch(&gates[(size_t)((b0 + prow) * T + t + 1) * GATE4 + pseg], 0, 3);
    }

    v8f acc[4];
    #pragma unroll
    for (int nt = 0; nt < 4; nt++) {               // C init = precomputed input gates
      int col = wave * 64 + nt * 16 + l15;
      #pragma unroll
      for (int r = 0; r < 8; r++) {
        int brow = r + 8 * hi;
        acc[nt][r] = gates[(size_t)((b0 + brow) * T + t) * GATE4 + col];
      }
    }
    #pragma unroll
    for (int kk = 0; kk < 4; kk++) {               // K = 128 in 4 steps of 32
      union { int i[8]; v16bf v; } a;
      #pragma unroll
      for (int q = 0; q < 8; q++) {
        int k0 = kk * 32 + 2 * q + hi * 16;
        a.i[q] = *(const int*)&h_s[l15][k0];
      }
      #pragma unroll
      for (int nt = 0; nt < 4; nt++) {
        acc[nt] = __builtin_amdgcn_wmma_f32_16x16x32_bf16(
            false, a.v, false, bfr[kk][nt].v, (short)0, acc[nt], false, false);
      }
    }
    #pragma unroll
    for (int nt = 0; nt < 4; nt++) {               // spill gate preacts to LDS
      int col = wave * 64 + nt * 16 + l15;
      #pragma unroll
      for (int r = 0; r < 8; r++) gbuf[r + 8 * hi][col] = acc[nt][r];
    }
    __syncthreads();

    for (int u = tid; u < 16 * 128; u += 256) {    // elementwise cell update
      int row = u >> 7, j = u & 127;
      float ig = gbuf[row][j];
      float fg = gbuf[row][128 + j];
      float gg = gbuf[row][256 + j];
      float og = gbuf[row][384 + j];
      float c  = c_s[row][j];
      float si = 1.0f / (1.0f + __expf(-ig));
      float sf = 1.0f / (1.0f + __expf(-fg));
      float so = 1.0f / (1.0f + __expf(-og));
      c = sf * c + si * tanhf(gg);
      float h = so * tanhf(c);
      c_s[row][j] = c;
      h_s[row][j] = f2bf(h);
      if (ys) ys[((size_t)(b0 + row) * T + t) * 128 + j] = f2bf(h);
      if (hT && t == T - 1) hT[(b0 + row) * 128 + j] = h;
    }
    __syncthreads();
  }
}

// ---------- host orchestration ----------
extern "C" void kernel_launch(void* const* d_in, const int* in_sizes, int n_in,
                              void* d_out, int out_size, void* d_ws, size_t ws_size,
                              hipStream_t stream) {
  const float* x     = (const float*)d_in[0];
  const int*   ei    = (const int*)d_in[1];
  const float* W0    = (const float*)d_in[3];
  const float* b0    = (const float*)d_in[4];
  const float* W1    = (const float*)d_in[5];
  const float* b1    = (const float*)d_in[6];
  const float* gamma = (const float*)d_in[7];
  const float* beta  = (const float*)d_in[8];
  const float* Wih0  = (const float*)d_in[9];
  const float* Whh0  = (const float*)d_in[10];
  const float* bih0  = (const float*)d_in[11];
  const float* bhh0  = (const float*)d_in[12];
  const float* Wih1  = (const float*)d_in[13];
  const float* Whh1  = (const float*)d_in[14];
  const float* bih1  = (const float*)d_in[15];
  const float* bhh1  = (const float*)d_in[16];
  const int* src = ei;
  const int* dst = ei + N_EDGES;

  char* ws = (char*)d_ws;
  size_t off = 0;
  auto carve = [&](size_t bytes) -> char* {
    char* p = ws + off;
    off += (bytes + 255) & ~(size_t)255;
    return p;
  };
  float* dinv   = (float*)carve((size_t)N_NODES * 4);
  float* nrm    = (float*)carve((size_t)N_EDGES * 4);
  float* h1     = (float*)carve((size_t)N_NODES * HID * 4);
  float* h2     = (float*)carve((size_t)N_NODES * HID * 4);
  u16*   actbf  = (u16*)  carve((size_t)N_NODES * HID * 2);
  u16*   xbf    = (u16*)  carve((size_t)N_NODES * IN_DIM * 2);
  u16*   w0t    = (u16*)  carve((size_t)HID * IN_DIM * 2);
  u16*   w1t    = (u16*)  carve((size_t)HID * HID * 2);
  u16*   wih0b  = (u16*)  carve((size_t)GATE4 * HID * 2);
  u16*   whh0b  = (u16*)  carve((size_t)GATE4 * OUT_DIM * 2);
  u16*   wih1b  = (u16*)  carve((size_t)GATE4 * OUT_DIM * 2);
  u16*   whh1b  = (u16*)  carve((size_t)GATE4 * OUT_DIM * 2);
  float* gb0    = (float*)carve((size_t)GATE4 * 4);
  float* gb1    = (float*)carve((size_t)GATE4 * 4);
  float* gates  = (float*)carve((size_t)N_NODES * GATE4 * 4);
  u16*   ys0    = (u16*)  carve((size_t)N_NODES * OUT_DIM * 2);

  const int TPB = 256;
  auto blk = [](long long n) { return (unsigned)((n + 255) / 256); };

  // --- GCN normalization ---
  k_fill1<<<blk(N_NODES), TPB, 0, stream>>>(dinv, N_NODES);            // deg=1 (self loop)
  k_deg<<<blk(N_EDGES), TPB, 0, stream>>>(dst, dinv, N_EDGES);
  k_rsqrt_inplace<<<blk(N_NODES), TPB, 0, stream>>>(dinv, N_NODES);
  k_norm<<<blk(N_EDGES), TPB, 0, stream>>>(src, dst, dinv, nrm, N_EDGES);

  // --- weight / input conversion to bf16 [N,K] layouts ---
  k_f2bf<<<blk((long long)N_NODES * IN_DIM), TPB, 0, stream>>>(x, xbf, N_NODES * IN_DIM);
  k_transpose_bf<<<blk(IN_DIM * HID), TPB, 0, stream>>>(W0, w0t, IN_DIM, HID);
  k_transpose_bf<<<blk(HID * HID), TPB, 0, stream>>>(W1, w1t, HID, HID);
  k_f2bf<<<blk(GATE4 * HID), TPB, 0, stream>>>(Wih0, wih0b, GATE4 * HID);
  k_f2bf<<<blk(GATE4 * OUT_DIM), TPB, 0, stream>>>(Whh0, whh0b, GATE4 * OUT_DIM);
  k_f2bf<<<blk(GATE4 * OUT_DIM), TPB, 0, stream>>>(Wih1, wih1b, GATE4 * OUT_DIM);
  k_f2bf<<<blk(GATE4 * OUT_DIM), TPB, 0, stream>>>(Whh1, whh1b, GATE4 * OUT_DIM);
  k_addbias2<<<blk(GATE4), TPB, 0, stream>>>(bih0, bhh0, gb0, GATE4);
  k_addbias2<<<blk(GATE4), TPB, 0, stream>>>(bih1, bhh1, gb1, GATE4);

  // --- GCN layer 0 ---
  gemm_bf16_nt<<<dim3(N_NODES / 128, HID / 128), TPB, 0, stream>>>(
      xbf, w0t, nullptr, h1, N_NODES, HID, IN_DIM);
  k_initagg<<<blk((long long)N_NODES * HID), TPB, 0, stream>>>(h1, dinv, b0, h2,
                                                               N_NODES * HID);
  k_scatter<<<blk((long long)N_EDGES * 64), TPB, 0, stream>>>(src, dst, nrm, h1, h2,
                                                              N_EDGES);
  k_relubf<<<blk((long long)N_NODES * HID), TPB, 0, stream>>>(h2, actbf, N_NODES * HID);

  // --- GCN layer 1 ---
  gemm_bf16_nt<<<dim3(N_NODES / 128, HID / 128), TPB, 0, stream>>>(
      actbf, w1t, nullptr, h1, N_NODES, HID, HID);
  k_initagg<<<blk((long long)N_NODES * HID), TPB, 0, stream>>>(h1, dinv, b1, h2,
                                                               N_NODES * HID);
  k_scatter<<<blk((long long)N_EDGES * 64), TPB, 0, stream>>>(src, dst, nrm, h1, h2,
                                                              N_EDGES);

  // --- LayerNorm -> bf16 sequence ---
  k_layernorm<<<N_NODES / 8, TPB, 0, stream>>>(h2, gamma, beta, actbf, N_NODES);

  // --- LSTM layer 0: big input-gate GEMM, then recurrence ---
  gemm_bf16_nt<<<dim3(N_NODES / 128, GATE4 / 128), TPB, 0, stream>>>(
      actbf, wih0b, gb0, gates, N_NODES, GATE4, HID);
  k_lstm<<<BATCH / 16, TPB, 0, stream>>>(gates, whh0b, ys0, nullptr, SEQ_T);

  // --- LSTM layer 1 ---
  gemm_bf16_nt<<<dim3(N_NODES / 128, GATE4 / 128), TPB, 0, stream>>>(
      ys0, wih1b, gb1, gates, N_NODES, GATE4, OUT_DIM);
  k_lstm<<<BATCH / 16, TPB, 0, stream>>>(gates, whh1b, nullptr, (float*)d_out, SEQ_T);
}
